// Multiheadselfattention_49984829390950
// MI455X (gfx1250) — matrix operand order repaired and checked by reading, not compiled, not run on hardware
//
#include <hip/hip_runtime.h>

typedef __attribute__((ext_vector_type(16))) _Float16 v16h;
typedef __attribute__((ext_vector_type(8)))  _Float16 v8h;
typedef __attribute__((ext_vector_type(8)))  float    v8f;

#define B_   8
#define N_   1024
#define TD   768
#define H_   12
#define D_   64
#define QKVN 2304

static __device__ __forceinline__ v8h ld8(const _Float16* p) {
    return *(const v8h*)p;
}
static __device__ __forceinline__ v16h ld16(const _Float16* p) {
    return *(const v16h*)p;
}
static __device__ __forceinline__ v16h mk16(v8h a, v8h b) {
    v16h r;
#pragma unroll
    for (int i = 0; i < 8; ++i) { r[i] = a[i]; r[i + 8] = b[i]; }
    return r;
}
static __device__ __forceinline__ v8f wmma32(v16h a, v16h b, v8f c) {
    // D(16x16,f32) = A(16x32,f16) * B(32x16,f16) + C
    return __builtin_amdgcn_wmma_f32_16x16x32_f16(false, a, false, b,
                                                  (short)0, c, false, false);
}
// Async DMA: global memory -> LDS (16 bytes per lane), tracked by ASYNCcnt.
static __device__ __forceinline__ void async_ld16(void* lds, const void* g) {
    uint32_t l = (uint32_t)(uintptr_t)lds;            // flat low 32 = LDS offset
    asm volatile("global_load_async_to_lds_b128 %0, %1, off"
                 :: "v"(l), "v"((uint64_t)(uintptr_t)g)
                 : "memory");
}

// ---------------------------------------------------------------------------
// Kernel 1: qkv = x @ W_qkv  (M=8192, K=768, N=2304), scatter to Qh/Kh/Vt f16
// using the reference's reshape(b,3,H,n,D) flat-index mapping. Branchless
// epilogue; register-double-buffered global->LDS staging.
// ---------------------------------------------------------------------------
__global__ __launch_bounds__(256) void qkv_gemm(const float* __restrict__ x,
                                                const float* __restrict__ Wqkv,
                                                _Float16* __restrict__ Qh,
                                                _Float16* __restrict__ Kh,
                                                _Float16* __restrict__ Vt) {
    __shared__ __align__(32) _Float16 aLds[128 * 32];
    __shared__ __align__(32) _Float16 bLds[128 * 32];
    const int t    = threadIdx.x;
    const int lane = t & 31, wid = t >> 5;
    const int hs = lane >> 4, lc = lane & 15;
    const int m0 = blockIdx.y * 128;
    const int n0 = blockIdx.x * 128;
    const int wm = (wid >> 1) * 32;
    const int wn = (wid & 1) * 64;

    v8f acc[2][4] = {};
    float4 aReg[4], bReg[4];

    auto gload = [&](int k0) {
#pragma unroll
        for (int i = 0; i < 4; ++i) {
            int id = t * 4 + i;            // 0..1023
            int r  = id >> 3;              // A row 0..127
            int q  = id & 7;               // float4 idx in row
            aReg[i] = *(const float4*)(x + (size_t)(m0 + r) * TD + k0 + q * 4);
        }
#pragma unroll
        for (int i = 0; i < 4; ++i) {
            int id = t * 4 + i;
            int r  = id >> 5;              // B k-row 0..31
            int q  = id & 31;              // float4 along n
            bReg[i] = *(const float4*)(Wqkv + (size_t)(k0 + r) * QKVN + n0 + q * 4);
        }
    };
    auto stage = [&]() {
#pragma unroll
        for (int i = 0; i < 4; ++i) {
            int id = t * 4 + i;
            int r  = id >> 3, q = id & 7;
            _Float16* dst = aLds + r * 32 + q * 4;
            dst[0] = (_Float16)aReg[i].x; dst[1] = (_Float16)aReg[i].y;
            dst[2] = (_Float16)aReg[i].z; dst[3] = (_Float16)aReg[i].w;
        }
#pragma unroll
        for (int i = 0; i < 4; ++i) {
            int id = t * 4 + i;
            int r  = id >> 5, q = id & 31;
            bLds[(q * 4 + 0) * 32 + r] = (_Float16)bReg[i].x;
            bLds[(q * 4 + 1) * 32 + r] = (_Float16)bReg[i].y;
            bLds[(q * 4 + 2) * 32 + r] = (_Float16)bReg[i].z;
            bLds[(q * 4 + 3) * 32 + r] = (_Float16)bReg[i].w;
        }
    };

    gload(0);
    for (int k0 = 0; k0 < TD; k0 += 32) {
        stage();
        __syncthreads();
        if (k0 + 32 < TD) gload(k0 + 32);   // prefetch next tile (overlaps WMMA)

        v16h aF[2], bF[4];
#pragma unroll
        for (int mt = 0; mt < 2; ++mt) {
            const _Float16* ap = aLds + (wm + mt * 16 + lc) * 32;
            aF[mt] = ld16(ap + hs * 16);
        }
#pragma unroll
        for (int nt = 0; nt < 4; ++nt)
            bF[nt] = ld16(bLds + (wn + nt * 16 + lc) * 32 + hs * 16);
#pragma unroll
        for (int mt = 0; mt < 2; ++mt)
#pragma unroll
            for (int nt = 0; nt < 4; ++nt)
                acc[mt][nt] = wmma32(aF[mt], bF[nt], acc[mt][nt]);
        __syncthreads();
    }

    // epilogue: reference reshape(b,3,H,n,D): flat = ni*2304 + c,
    // d = flat&63, n2 = (flat>>6)&1023, hh = flat>>16 in [0,36) = s*12+h.
    // Branchless pointer/index select — no exec-mask divergence.
#pragma unroll
    for (int mt = 0; mt < 2; ++mt)
#pragma unroll
        for (int nt = 0; nt < 4; ++nt)
#pragma unroll
            for (int r = 0; r < 8; ++r) {
                int m  = m0 + wm + mt * 16 + r + 8 * hs;
                int c  = n0 + wn + nt * 16 + lc;
                int b  = m >> 10, ni = m & 1023;
                int flat = ni * QKVN + c;
                int d  = flat & 63;
                int t2 = flat >> 6;
                int n2 = t2 & 1023;
                int hh = t2 >> 10;                          // 0..35
                int s  = (hh >= 24) ? 2 : ((hh >= 12) ? 1 : 0);
                int bh = b * H_ + (hh - s * 12);
                size_t idxQK = ((size_t)bh << 16) + (n2 << 6) + d;
                size_t idxV  = ((((size_t)bh << 6) + d) << 10) + n2;
                _Float16* p = (s == 2) ? (Vt + idxV)
                                       : (((s == 1) ? Kh : Qh) + idxQK);
                *p = (_Float16)acc[mt][nt][r];
            }
}

// ---------------------------------------------------------------------------
// Kernel 2: flash attention per (bh, 128-query block). 8 waves, 16 rows each.
// K/V blocks DMA'd via GLOBAL_LOAD_ASYNC_TO_LDS with double-buffered LDS.
// ---------------------------------------------------------------------------
__global__ __launch_bounds__(256) void attn_kernel(const _Float16* __restrict__ Qh,
                                                   const _Float16* __restrict__ Kh,
                                                   const _Float16* __restrict__ Vt,
                                                   _Float16* __restrict__ attn) {
    __shared__ __align__(32) _Float16 kLds[2][32 * 64];   // [buf][key][d]
    __shared__ __align__(32) _Float16 vLds[2][64 * 32];   // [buf][d][key]
    __shared__ __align__(32) _Float16 pLds[8 * 16 * 32];  // per-wave P patch
    const int t    = threadIdx.x;
    const int lane = t & 31, wid = t >> 5;
    const int hs = lane >> 4, lc = lane & 15;
    const int bh = blockIdx.y;                 // 0..95
    const int q0 = blockIdx.x * 128 + wid * 16;

    // Q fragments for this wave's 16 rows, pre-scaled by 1/sqrt(D)=0.125
    v16h qF[2];
    {
        const _Float16* qp = Qh + ((size_t)bh << 16) + ((size_t)(q0 + lc) << 6);
#pragma unroll
        for (int ks = 0; ks < 2; ++ks) {
            v8h a = ld8(qp + ks * 32 + hs * 16);
            v8h b = ld8(qp + ks * 32 + hs * 16 + 8);
#pragma unroll
            for (int i = 0; i < 8; ++i) {
                a[i] = a[i] * (_Float16)0.125f;
                b[i] = b[i] * (_Float16)0.125f;
            }
            qF[ks] = mk16(a, b);
        }
    }

    float mrow[8], lrow[8];
    v8f acc[4] = {};
#pragma unroll
    for (int r = 0; r < 8; ++r) { mrow[r] = -1e30f; lrow[r] = 0.0f; }

    // async K/V staging: 16 bytes of K + 16 bytes of V per thread per block
    const int kkey = t >> 3, kch = t & 7;   // K: key row, 8-half chunk
    const int vd   = t >> 2, vch = t & 3;   // V: d row, 8-half chunk
    const _Float16* kBase = Kh + ((size_t)bh << 16);
    const _Float16* vBase = Vt + (((size_t)bh << 6)) * (size_t)N_;
    auto issue = [&](int kb, int buf) {
        async_ld16(&kLds[buf][kkey * 64 + kch * 8],
                   kBase + ((size_t)(kb + kkey) << 6) + kch * 8);
        async_ld16(&vLds[buf][vd * 32 + vch * 8],
                   vBase + (size_t)vd * N_ + kb + vch * 8);
    };

    issue(0, 0);
    int cur = 0;
    for (int kb = 0; kb < N_; kb += 32, cur ^= 1) {
        if (kb + 32 < N_) {
            issue(kb + 32, cur ^ 1);                       // overlap DMA w/ compute
            asm volatile("s_wait_asynccnt 0x2" ::: "memory");
        } else {
            asm volatile("s_wait_asynccnt 0x0" ::: "memory");
        }
        __syncthreads();
        const _Float16* kl = kLds[cur];
        const _Float16* vl = vLds[cur];

        // S = Q K^T for 16 rows x 32 keys: two 16x16 tiles, 2 K-steps each
        v8f s0 = {}, s1 = {};
        {
            const _Float16* kp0 = kl + lc * 64;            // key tile 0
            const _Float16* kp1 = kl + (16 + lc) * 64;     // key tile 1
            s0 = wmma32(qF[0], ld16(kp0 + hs * 16), s0);
            s0 = wmma32(qF[1], ld16(kp0 + 32 + hs * 16), s0);
            s1 = wmma32(qF[0], ld16(kp1 + hs * 16), s1);
            s1 = wmma32(qF[1], ld16(kp1 + 32 + hs * 16), s1);
        }

        // online softmax (row lives across 16 lanes of a half-wave)
        float p0[8], p1[8];
#pragma unroll
        for (int r = 0; r < 8; ++r) {
            float mx = fmaxf(s0[r], s1[r]);
#pragma unroll
            for (int msk = 1; msk < 16; msk <<= 1)
                mx = fmaxf(mx, __shfl_xor(mx, msk, 32));
            float mn = fmaxf(mrow[r], mx);
            float al = __expf(mrow[r] - mn);
            mrow[r] = mn;
            p0[r] = __expf(s0[r] - mn);
            p1[r] = __expf(s1[r] - mn);
            float sm = p0[r] + p1[r];
#pragma unroll
            for (int msk = 1; msk < 16; msk <<= 1)
                sm += __shfl_xor(sm, msk, 32);
            lrow[r] = lrow[r] * al + sm;
#pragma unroll
            for (int dt = 0; dt < 4; ++dt) acc[dt][r] *= al;
        }

        // re-layout P (C-layout) -> A-fragment via per-wave LDS patch
        _Float16* pw = pLds + wid * 512;
#pragma unroll
        for (int r = 0; r < 8; ++r) {
            int row = r + 8 * hs;
            pw[row * 32 + lc]      = (_Float16)p0[r];
            pw[row * 32 + 16 + lc] = (_Float16)p1[r];
        }
        asm volatile("s_wait_dscnt 0" ::: "memory");   // wave-local LDS RAW
        const _Float16* pr = pw + lc * 32;
        v16h pF = ld16(pr + hs * 16);

        // acc += P @ V  (contract over 32 keys; 4 d-tiles)
#pragma unroll
        for (int dt = 0; dt < 4; ++dt) {
            v16h vF = ld16(vl + (dt * 16 + lc) * 32 + hs * 16);
            acc[dt] = wmma32(pF, vF, acc[dt]);
        }
        __syncthreads();   // all waves done with buf[cur] before it is re-DMA'd
    }

    // epilogue: attn[b, n, h*64+d] = acc / l  (f16)
    int b = bh / H_, h = bh % H_;
#pragma unroll
    for (int r = 0; r < 8; ++r) {
        float rl = 1.0f / lrow[r];
        int row = q0 + r + 8 * hs;
        size_t base = ((size_t)(b * N_ + row)) * TD + h * D_;
#pragma unroll
        for (int dt = 0; dt < 4; ++dt)
            attn[base + dt * 16 + lc] = (_Float16)(acc[dt][r] * rl);
    }
}

// ---------------------------------------------------------------------------
// Kernel 3: out = attn @ W_out + b_out  (M=8192, K=768, N=768), f32 output
// ---------------------------------------------------------------------------
__global__ __launch_bounds__(256) void out_gemm(const _Float16* __restrict__ attn,
                                                const float* __restrict__ Wout,
                                                const float* __restrict__ bout,
                                                float* __restrict__ out) {
    __shared__ __align__(32) _Float16 aLds[128 * 32];
    __shared__ __align__(32) _Float16 bLds[128 * 32];
    const int t    = threadIdx.x;
    const int lane = t & 31, wid = t >> 5;
    const int hs = lane >> 4, lc = lane & 15;
    const int m0 = blockIdx.y * 128;
    const int n0 = blockIdx.x * 128;
    const int wm = (wid >> 1) * 32;
    const int wn = (wid & 1) * 64;

    v8f acc[2][4] = {};
    v8h aReg[2];
    float4 bReg[4];

    auto gload = [&](int k0) {
#pragma unroll
        for (int i = 0; i < 2; ++i) {
            int id = t * 2 + i;            // 0..511
            int r  = id >> 2;              // A row 0..127
            int ch = id & 3;               // 8-half chunk
            aReg[i] = ld8(attn + (size_t)(m0 + r) * TD + k0 + ch * 8);
        }
#pragma unroll
        for (int i = 0; i < 4; ++i) {
            int id = t * 4 + i;
            int r  = id >> 5;
            int q  = id & 31;
            bReg[i] = *(const float4*)(Wout + (size_t)(k0 + r) * TD + n0 + q * 4);
        }
    };
    auto stage = [&]() {
#pragma unroll
        for (int i = 0; i < 2; ++i) {
            int id = t * 2 + i;
            int r  = id >> 2, ch = id & 3;
            *(v8h*)(aLds + r * 32 + ch * 8) = aReg[i];
        }
#pragma unroll
        for (int i = 0; i < 4; ++i) {
            int id = t * 4 + i;
            int r  = id >> 5, q = id & 31;
            bLds[(q * 4 + 0) * 32 + r] = (_Float16)bReg[i].x;
            bLds[(q * 4 + 1) * 32 + r] = (_Float16)bReg[i].y;
            bLds[(q * 4 + 2) * 32 + r] = (_Float16)bReg[i].z;
            bLds[(q * 4 + 3) * 32 + r] = (_Float16)bReg[i].w;
        }
    };

    gload(0);
    for (int k0 = 0; k0 < TD; k0 += 32) {
        stage();
        __syncthreads();
        if (k0 + 32 < TD) gload(k0 + 32);   // prefetch next tile

        v16h aF[2], bF[4];
#pragma unroll
        for (int mt = 0; mt < 2; ++mt) {
            const _Float16* ap = aLds + (wm + mt * 16 + lc) * 32;
            aF[mt] = ld16(ap + hs * 16);
        }
#pragma unroll
        for (int nt = 0; nt < 4; ++nt)
            bF[nt] = ld16(bLds + (wn + nt * 16 + lc) * 32 + hs * 16);
#pragma unroll
        for (int mt = 0; mt < 2; ++mt)
#pragma unroll
            for (int nt = 0; nt < 4; ++nt)
                acc[mt][nt] = wmma32(aF[mt], bF[nt], acc[mt][nt]);
        __syncthreads();
    }

#pragma unroll
    for (int mt = 0; mt < 2; ++mt)
#pragma unroll
        for (int nt = 0; nt < 4; ++nt)
#pragma unroll
            for (int r = 0; r < 8; ++r) {
                int m = m0 + wm + mt * 16 + r + 8 * hs;
                int c = n0 + wn + nt * 16 + lc;
                out[(size_t)m * TD + c] = acc[mt][nt][r] + bout[c];
            }
}

// ---------------------------------------------------------------------------
extern "C" void kernel_launch(void* const* d_in, const int* in_sizes, int n_in,
                              void* d_out, int out_size, void* d_ws, size_t ws_size,
                              hipStream_t stream) {
    const float* x    = (const float*)d_in[0];
    const float* Wqkv = (const float*)d_in[1];
    const float* Wout = (const float*)d_in[2];
    const float* bout = (const float*)d_in[3];
    float* out = (float*)d_out;

    const size_t SZ = (size_t)B_ * H_ * N_ * D_;   // 6,291,456 elems
    _Float16* Qh    = (_Float16*)d_ws;
    _Float16* Kh    = Qh + SZ;
    _Float16* Vt    = Kh + SZ;
    _Float16* attnb = Vt + SZ;                     // B*N x 768

    qkv_gemm  <<<dim3(QKVN / 128, (B_ * N_) / 128), 256, 0, stream>>>(x, Wqkv, Qh, Kh, Vt);
    attn_kernel<<<dim3(N_ / 128, B_ * H_),          256, 0, stream>>>(Qh, Kh, Vt, attnb);
    out_gemm  <<<dim3(TD / 128, (B_ * N_) / 128),   256, 0, stream>>>(attnb, Wout, bout, out);
}
